// RasterCaster_72258529788394
// MI455X (gfx1250) — matrix operand order repaired and checked by reading, not compiled, not run on hardware
//
#include <hip/hip_runtime.h>

// ---------------------------------------------------------------------------
// RasterCaster fused pipeline for gfx1250 (MI455X), bf16 WMMA 16x16x32.
// ---------------------------------------------------------------------------

#define NHID   128
#define NVROWS 100001            // NV + 1 embedding rows
#define NPIX   (4 * 128 * 128)   // N*H*W pixels

typedef __attribute__((ext_vector_type(16))) __bf16 v16bf;
typedef __attribute__((ext_vector_type(4)))  __bf16 v4bf;
typedef __attribute__((ext_vector_type(8)))  float  v8f;

union FragBF  { v16bf v; uint4 q[2]; };
union PackBF4 { v4bf v; uint2 q; };

// PE sigmas: 2*pi / 0.1^(2k/proj)
#define SIGE0 6.28318531f      // emb PE, k=0
#define SIGE1 19.8691765f      // emb PE, k=1 : 2pi*sqrt(10)
#define SIGV0 6.28318531f      // view PE, k=0
#define SIGV1 13.5366707f      // view PE, k=1 : 2pi*10^(1/3)
#define SIGV2 29.1637984f      // view PE, k=2 : 2pi*10^(2/3)

// ---------------------------------------------------------------------------
// Wave-level GEMM: 16 rows (this wave's tile) x (NT*16) cols, K = 32*ksteps.
// Activations: LDS row-major bf16 [16][actStride]. Weights: global bf16 W^T
// row-major [N][wtStride] (so each lane's B-fragment is contiguous in K).
// A layout (matches B): lane<16 -> K k0..k0+15,
// lane>=16 -> K k0+16..k0+31. B layout (lanes = N): lane<16 holds
// K k0..k0+15, lane>=16 holds K k0+16..k0+31.
// ---------------------------------------------------------------------------
template <int NT>
__device__ __forceinline__ void wave_gemm(const __bf16* act, int actStride,
                                          const __bf16* wt, int wtStride,
                                          int ksteps, v8f* acc) {
    const int lane = threadIdx.x & 31;
    const int l15  = lane & 15;
    const int half = lane >> 4;
    for (int ks = 0; ks < ksteps; ++ks) {
        const int k0 = ks * 32;
        FragBF a;
        const __bf16* ar = act + l15 * actStride + k0 + half * 16;
        a.q[0] = *(const uint4*)(ar);
        a.q[1] = *(const uint4*)(ar + 8);
#pragma unroll
        for (int nt = 0; nt < NT; ++nt) {
            FragBF b;
            const __bf16* br = wt + (nt * 16 + l15) * wtStride + k0 + half * 16;
            b.q[0] = *(const uint4*)(br);
            b.q[1] = *(const uint4*)(br + 8);
            acc[nt] = __builtin_amdgcn_wmma_f32_16x16x32_bf16(
                false, a.v, false, b.v, (short)0, acc[nt], false, false);
        }
    }
}

// ---------------------------------------------------------------------------
// Prep kernel 1: per-row inverse norm scale for nn.Embedding(max_norm=1.0)
// ---------------------------------------------------------------------------
__global__ void rc_invnorm(const float* __restrict__ emb, float* __restrict__ invn, int rows) {
    int r = blockIdx.x * blockDim.x + threadIdx.x;
    if (r >= rows) return;
    const float* p = emb + (long long)r * NHID;
    float s = 0.f;
    for (int i = 0; i < NHID; ++i) { float x = p[i]; s += x * x; }
    float n = sqrtf(s);
    invn[r] = (n > 1.0f) ? (1.0f / n) : 1.0f;   // min(1, 1/max(n,1e-12))
}

// ---------------------------------------------------------------------------
// Prep kernel 2: fp32 W[K][N] -> bf16 W^T[Np][Kp] (zero padded)
// ---------------------------------------------------------------------------
__global__ void rc_wt(const float* __restrict__ src, __bf16* __restrict__ dst,
                      int K, int Nn, int Kp, int Np) {
    int i = blockIdx.x * blockDim.x + threadIdx.x;
    if (i >= Kp * Np) return;
    int n = i / Kp, k = i % Kp;
    float v = (k < K && n < Nn) ? src[(long long)k * Nn + n] : 0.f;
    dst[i] = (__bf16)v;
}

// ---------------------------------------------------------------------------
// Fused main kernel: 4 waves/block, 16 pixels/wave.
// ---------------------------------------------------------------------------
struct RCParams {
    const int*   vidx;  const float* bary;  const float* view;
    const float* emb;   const float* invn;
    const __bf16* sinWt;  const float* sin_b;
    const __bf16* vdWt;   const float* vd_b;
    const __bf16* vdW1t;  const float* vdb1;
    const __bf16* vdW2t;  const float* vdb2;
    const __bf16* vdW3t;  const float* vdb3;
    const __bf16* hWt[12]; const float* hB[12];
    const __bf16* hd1t;   const float* hdb1;
    const __bf16* hd2t;   const float* hdb2;
    const __bf16* hd3t;   const float* hdb3;
    float* out;
};

__global__ __launch_bounds__(128, 1) void rc_main(RCParams P) {
    __shared__ __align__(16) __bf16 sA[4][16 * NHID];  // scratch / vd acts
    __shared__ __align__(16) __bf16 sH[4][16 * NHID];  // main activations
    __shared__ __align__(16) __bf16 sV[4][16 * NHID];  // final vd (bf16)
    __shared__ float sBsum[4][16];

    const int wave  = threadIdx.x >> 5;
    const int lane  = threadIdx.x & 31;
    const int col   = lane & 15;
    const int rbase = (lane >> 4) * 8;           // C-layout: row = rbase + vgpr
    const int pxbase = blockIdx.x * 64 + wave * 16;

    __bf16* bufA = sA[wave];
    __bf16* bufH = sH[wave];
    __bf16* bufV = sV[wave];
    float*  bsum = sBsum[wave];

    __builtin_prefetch(P.sinWt, 0, 1);           // global_prefetch_b8 into L2

    const v8f vzero = {0.f, 0.f, 0.f, 0.f, 0.f, 0.f, 0.f, 0.f};

    // Generic layer: D = act(src @ W^T_bf16 + bias) [*vd], written as bf16.
    auto layer = [&](const __bf16* wt, int wtStride, const float* bias,
                     int ksteps, bool relu, bool vdmul,
                     const __bf16* src, __bf16* dst) {
        v8f acc[8];
#pragma unroll
        for (int i = 0; i < 8; ++i) acc[i] = vzero;
        wave_gemm<8>(src, NHID, wt, wtStride, ksteps, acc);
        __syncthreads();
#pragma unroll
        for (int nt = 0; nt < 8; ++nt) {
            float bv = bias[nt * 16 + col];
#pragma unroll
            for (int v = 0; v < 8; ++v) {
                int row = rbase + v;
                float x = acc[nt][v] + bv;
                if (relu) x = fmaxf(x, 0.f);
                if (vdmul) x *= (float)bufV[row * NHID + nt * 16 + col];
                dst[row * NHID + nt * 16 + col] = (__bf16)x;
            }
        }
        __syncthreads();
    };

    // ---------------- view-direction branch ----------------
    for (int p = 0; p < 16; ++p) {               // PE(view_dir) -> 18 feats, pad to 32
        float r = 0.f;
        if (lane < 18) {
            int c = lane / 6, j = lane % 6;
            float x  = P.view[(pxbase + p) * 3 + c];
            float sg = (j % 3 == 0) ? SIGV0 : ((j % 3 == 1) ? SIGV1 : SIGV2);
            float t  = x * sg;
            r = (j < 3) ? __sinf(t) : __cosf(t);
        }
        bufA[p * NHID + lane] = (__bf16)r;
    }
    __syncthreads();
    layer(P.vdWt,  32,   P.vd_b, 1, false, false, bufA, bufA);   // 18(->32)x128
    layer(P.vdW1t, NHID, P.vdb1, 4, true,  false, bufA, bufA);
    layer(P.vdW2t, NHID, P.vdb2, 4, true,  false, bufA, bufA);
    layer(P.vdW3t, NHID, P.vdb3, 4, true,  false, bufA, bufV);   // outer relu folded

    // ---------------- gather + PE + sin GEMM (K = 512, chunked by 128) ------
    {
        v8f acc[8];
#pragma unroll
        for (int i = 0; i < 8; ++i) acc[i] = vzero;

        for (int kc = 0; kc < 4; ++kc) {
            const int h = kc * 32 + lane;        // lane -> one embedding element
            for (int p = 0; p < 16; ++p) {
                const int px = pxbase + p;
                float a0 = 0.f, a1 = 0.f, a2 = 0.f, a3 = 0.f, bs = 0.f;
                for (int v = 0; v < 3; ++v) {
                    const int   idx = P.vidx[px * 3 + v] + 1;
                    const float b   = P.bary[px * 3 + v];
                    const float x   = P.emb[(long long)idx * NHID + h] * P.invn[idx];
                    const float t0 = x * SIGE0, t1 = x * SIGE1;
                    a0 += b * __sinf(t0); a1 += b * __sinf(t1);
                    a2 += b * __cosf(t0); a3 += b * __cosf(t1);
                    bs += b;
                }
                // chunk-local feature cols: lane*4 + {0..3}  (k = h*4 + j)
                PackBF4 o;
                o.v = (v4bf){(__bf16)a0, (__bf16)a1, (__bf16)a2, (__bf16)a3};
                *(uint2*)(bufA + p * NHID + lane * 4) = o.q;
                if (kc == 0 && lane == 0) bsum[p] = bs;
            }
            __syncthreads();
            wave_gemm<8>(bufA, NHID, P.sinWt + kc * 128, 512, 4, acc);
            __syncthreads();                     // bufA rewritten next chunk
        }
        // epilogue: + sin_b[n] * (sum of barys), no relu
#pragma unroll
        for (int nt = 0; nt < 8; ++nt) {
            float bv = P.sin_b[nt * 16 + col];
#pragma unroll
            for (int v = 0; v < 8; ++v) {
                int row = rbase + v;
                bufH[row * NHID + nt * 16 + col] = (__bf16)(acc[nt][v] + bv * bsum[row]);
            }
        }
        __syncthreads();
    }

    // ---------------- hidden MLP chain ----------------
#pragma unroll 1
    for (int m = 0; m < 4; ++m) {
        layer(P.hWt[3 * m + 0], NHID, P.hB[3 * m + 0], 4, true, false,     bufH, bufH);
        layer(P.hWt[3 * m + 1], NHID, P.hB[3 * m + 1], 4, true, false,     bufH, bufH);
        layer(P.hWt[3 * m + 2], NHID, P.hB[3 * m + 2], 4, true, (m == 0),  bufH, bufH);
    }

    // ---------------- head ----------------
    layer(P.hd1t, NHID, P.hdb1, 4, true, false, bufH, bufH);
    layer(P.hd2t, NHID, P.hdb2, 4, true, false, bufH, bufH);
    {
        v8f acc1[1]; acc1[0] = vzero;
        wave_gemm<1>(bufH, NHID, P.hd3t, NHID, 4, acc1);   // N padded 3 -> 16
        if (col < 3) {
            float bv = P.hdb3[col];
#pragma unroll
            for (int v = 0; v < 8; ++v) {
                int row = rbase + v;
                P.out[(long long)(pxbase + row) * 3 + col] = acc1[0][v] + bv;
            }
        }
    }
}

// ---------------------------------------------------------------------------
// Host launcher.
// Input order (setup_inputs dict, params flattened in insertion order):
//  0 vert_idx(i32) 1 bary 2 view_dir 3 emb 4 sin_W 5 sin_b 6 vd_W 7 vd_b
//  8..13 vd_mlp[0]{w1,b1,w2,b2,w3,b3}  14..37 hiddens[0..3]{...}  38..43 head
// ---------------------------------------------------------------------------
extern "C" void kernel_launch(void* const* d_in, const int* in_sizes, int n_in,
                              void* d_out, int out_size, void* d_ws, size_t ws_size,
                              hipStream_t stream) {
    (void)in_sizes; (void)n_in; (void)out_size; (void)ws_size;

    const int*   vidx = (const int*)d_in[0];
    const float* bary = (const float*)d_in[1];
    const float* view = (const float*)d_in[2];
    const float* emb  = (const float*)d_in[3];

    // workspace carve-out (~1.1 MB total)
    char* ws = (char*)d_ws; size_t off = 0;
    auto carve = [&](size_t bytes) { char* p = ws + off; off = (off + bytes + 255) & ~(size_t)255; return p; };
    float*   invn  = (float*)carve(NVROWS * sizeof(float));
    __bf16*  sinWt = (__bf16*)carve(128 * 512 * 2);
    __bf16*  vdWt  = (__bf16*)carve(128 * 32 * 2);
    __bf16*  mat[17];
    for (int i = 0; i < 17; ++i) mat[i] = (__bf16*)carve(128 * 128 * 2);
    __bf16*  hd3t  = (__bf16*)carve(16 * 128 * 2);

    rc_invnorm<<<(NVROWS + 255) / 256, 256, 0, stream>>>(emb, invn, NVROWS);

    auto wt = [&](const void* src, __bf16* dst, int K, int Nn, int Kp, int Np) {
        int total = Kp * Np;
        rc_wt<<<(total + 255) / 256, 256, 0, stream>>>((const float*)src, dst, K, Nn, Kp, Np);
    };
    wt(d_in[4], sinWt, 512, 128, 512, 128);
    wt(d_in[6], vdWt,  18,  128, 32,  128);
    wt(d_in[8],  mat[0], 128, 128, 128, 128);
    wt(d_in[10], mat[1], 128, 128, 128, 128);
    wt(d_in[12], mat[2], 128, 128, 128, 128);
    for (int m = 0; m < 4; ++m) {
        wt(d_in[14 + 6 * m + 0], mat[3 + 3 * m + 0], 128, 128, 128, 128);
        wt(d_in[14 + 6 * m + 2], mat[3 + 3 * m + 1], 128, 128, 128, 128);
        wt(d_in[14 + 6 * m + 4], mat[3 + 3 * m + 2], 128, 128, 128, 128);
    }
    wt(d_in[38], mat[15], 128, 128, 128, 128);
    wt(d_in[40], mat[16], 128, 128, 128, 128);
    wt(d_in[42], hd3t,    128, 3,   128, 16);

    RCParams P;
    P.vidx = vidx; P.bary = bary; P.view = view; P.emb = emb; P.invn = invn;
    P.sinWt = sinWt; P.sin_b = (const float*)d_in[5];
    P.vdWt  = vdWt;  P.vd_b  = (const float*)d_in[7];
    P.vdW1t = mat[0]; P.vdb1 = (const float*)d_in[9];
    P.vdW2t = mat[1]; P.vdb2 = (const float*)d_in[11];
    P.vdW3t = mat[2]; P.vdb3 = (const float*)d_in[13];
    for (int m = 0; m < 4; ++m) {
        P.hWt[3 * m + 0] = mat[3 + 3 * m + 0]; P.hB[3 * m + 0] = (const float*)d_in[15 + 6 * m];
        P.hWt[3 * m + 1] = mat[3 + 3 * m + 1]; P.hB[3 * m + 1] = (const float*)d_in[17 + 6 * m];
        P.hWt[3 * m + 2] = mat[3 + 3 * m + 2]; P.hB[3 * m + 2] = (const float*)d_in[19 + 6 * m];
    }
    P.hd1t = mat[15]; P.hdb1 = (const float*)d_in[39];
    P.hd2t = mat[16]; P.hdb2 = (const float*)d_in[41];
    P.hd3t = hd3t;    P.hdb3 = (const float*)d_in[43];
    P.out  = (float*)d_out;

    rc_main<<<NPIX / 64, 128, 0, stream>>>(P);
}
